// GraphLogitLayers_29489245455003
// MI455X (gfx1250) — matrix operand order repaired and checked by reading, not compiled, not run on hardware
//
#include <hip/hip_runtime.h>
#include <hip/hip_bf16.h>

// ---------------------------------------------------------------------------
// Problem constants (from reference): P=4, B=128, C=32, H=W=64, HID=64
// Conv: 3x3 circular, 32->64 ch, ReLU, global mean pool  -> nodes [B,P,64]
// Graph MLP: edges(128)->64->64, mean over send axis, 64->64->1 -> out [B,P]
// ---------------------------------------------------------------------------

#define PP   4
#define BB   128
#define CC   32
#define HH   64
#define WW   64
#define HID  64
#define NIMG (PP * BB)          // 512 images

typedef __attribute__((ext_vector_type(16))) __bf16 v16bf;
typedef __attribute__((ext_vector_type(8)))  float  v8f;

union AFrag { uint4 q[2]; v16bf v; };           // 32 bytes: A 16x32 bf16 fragment (per lane)
union BFrag { unsigned short u[16]; v16bf v; }; // 32 bytes: B 32x16 bf16 fragment (per lane)

static __device__ __forceinline__ unsigned short f2bf_rne(float f) {
    unsigned u = __float_as_uint(f);
    unsigned r = u + 0x7fffu + ((u >> 16) & 1u);   // round-to-nearest-even
    return (unsigned short)(r >> 16);
}

// ---------------------------------------------------------------------------
// Kernel 1: implicit-GEMM conv (bf16 WMMA, f32 accum) + bias + ReLU + pool.
// One workgroup per image. 256 threads = 8 waves.
//   wave w: ntile = w&3 (out-ch group of 16), mhalf = w>>2 (row half of band)
// LDS: band of 10 rows (8 computed + circular halo) as bf16 [yy][x][c]
//      addr(yy,x,c) = yy*4096 + x*64 + c*2   (40960 B) + 64-float pool area.
// ---------------------------------------------------------------------------
#define BAND_ROWS   8
#define STAGE_ROWS  10
#define LDS_IN_B    (STAGE_ROWS * WW * CC * 2)   // 40960
#define POOL_OFF    LDS_IN_B

__global__ __launch_bounds__(256)
void conv_pool_kernel(const float* __restrict__ x,
                      const float* __restrict__ conv_w,
                      const float* __restrict__ conv_b,
                      float* __restrict__ nodes)   // [B][P][HID]
{
    __shared__ __align__(16) unsigned char smem[LDS_IN_B + HID * 4];

    const int tid   = threadIdx.x;
    const int lane  = tid & 31;
    const int wave  = tid >> 5;
    const int ntile = wave & 3;     // output-channel tile (16 ch)
    const int mhalf = wave >> 1 >> 1; // wave>>2
    const int n     = blockIdx.x;   // image id: p = n>>7, b = n&127
    const float* xin = x + (size_t)n * (CC * HH * WW);

    // ---- build resident B fragments: B[k=c][n=oc] per 3x3 tap -------------
    // B k order used here (contiguous, matches A): lanes 0-15 hold K=0..15, lanes 16-31 hold K=16..31,
    // VGPR v holds K = khalf*16 + 2v (lo), +1 (hi); column n = lane&15.
    const int khalf = lane >> 4;
    const int nn    = lane & 15;
    const int oc    = ntile * 16 + nn;
    BFrag bfrag[9];
#pragma unroll
    for (int t = 0; t < 9; ++t) {
        const int ty = t / 3, tx = t % 3;
#pragma unroll
        for (int v = 0; v < 8; ++v) {
            const int c0 = khalf * 16 + 2 * v;
            bfrag[t].u[2 * v]     = f2bf_rne(conv_w[((oc * CC + c0)     * 3 + ty) * 3 + tx]);
            bfrag[t].u[2 * v + 1] = f2bf_rne(conv_w[((oc * CC + c0 + 1) * 3 + ty) * 3 + tx]);
        }
    }
    const float bias = conv_b[oc];

    if (tid < HID) ((float*)(smem + POOL_OFF))[tid] = 0.0f;

    float pool = 0.0f;

    for (int band = 0; band < HH / BAND_ROWS; ++band) {
        const int r0 = band * BAND_ROWS;

        // ---- stage band (f32 global -> bf16 LDS, transpose c to innermost)
        // 10 rows x 32 ch x 16 float4 = 5120 float4 loads
        for (int it = tid; it < STAGE_ROWS * CC * (WW / 4); it += 256) {
            const int xq = it & 15;
            const int c  = (it >> 4) & 31;
            const int yy = it >> 9;                     // 0..9
            const int gy = (r0 + yy + (HH - 1)) & (HH - 1);   // r0-1+yy mod 64
            const float4 f4 = *(const float4*)(xin + (c * HH + gy) * WW + xq * 4);
            unsigned lb = (unsigned)(yy * 4096 + xq * 256 + c * 2);
            *(unsigned short*)(smem + lb)       = f2bf_rne(f4.x);
            *(unsigned short*)(smem + lb + 64)  = f2bf_rne(f4.y);
            *(unsigned short*)(smem + lb + 128) = f2bf_rne(f4.z);
            *(unsigned short*)(smem + lb + 192) = f2bf_rne(f4.w);
            // prefetch next band's rows into cache (global_prefetch_b8)
            if (band < HH / BAND_ROWS - 1 && (it & 3) == 0) {
                const int gyn = (r0 + BAND_ROWS + yy + (HH - 1)) & (HH - 1);
                __builtin_prefetch(xin + (c * HH + gyn) * WW + xq * 4, 0, 1);
            }
        }
        __syncthreads();

        // ---- compute: 16 tiles per wave (4 rows x 4 x-tiles) ---------------
        const int m = lane & 15;                 // pixel within M tile
        const int csel = (lane >> 4) << 5;       // A: lanes>=16 take channels +16
        for (int t16 = 0; t16 < 16; ++t16) {
            const int lr = mhalf * 4 + (t16 >> 2);   // computed row in band
            const int x0 = (t16 & 3) * 16;
            v8f acc = {};
#pragma unroll
            for (int dyi = 0; dyi < 3; ++dyi) {
                const int yy = lr + dyi;             // staged row (halo at yy=0)
#pragma unroll
                for (int dxi = 0; dxi < 3; ++dxi) {
                    const int xx = (x0 + m + dxi + (WW - 1)) & (WW - 1);
                    const unsigned a_off = (unsigned)(yy * 4096 + xx * 64 + csel);
                    AFrag a;
                    a.q[0] = *(const uint4*)(smem + a_off);        // K 0..7 / 16..23
                    a.q[1] = *(const uint4*)(smem + a_off + 16);   // K 8..15 / 24..31
                    acc = __builtin_amdgcn_wmma_f32_16x16x32_bf16(
                        false, a.v, false, bfrag[dyi * 3 + dxi].v,
                        (short)0, acc, false, false);
                }
            }
            // bias + ReLU + pool partial sum (sum over all M is layout-free)
#pragma unroll
            for (int r = 0; r < 8; ++r)
                pool += fmaxf(acc[r] + bias, 0.0f);
        }
        __syncthreads();   // WAR: before next band's staging
    }

    // ---- reduce pool across waves/lanes in LDS, write nodes ---------------
    atomicAdd((float*)(smem + POOL_OFF) + (ntile * 16 + (lane & 15)), pool);
    __syncthreads();
    if (tid < HID) {
        const int p = n >> 7, b = n & 127;
        nodes[((size_t)b * PP + p) * HID + tid] =
            ((float*)(smem + POOL_OFF))[tid] * (1.0f / (HH * WW));
    }
}

// ---------------------------------------------------------------------------
// Kernel 2: tiny graph MLP (f32 VALU — <0.1% of FLOPs, latency bound).
// One block per batch element b.
// ---------------------------------------------------------------------------
__global__ __launch_bounds__(256)
void graph_mlp_kernel(const float* __restrict__ nodes,   // [B][P][HID]
                      const float* __restrict__ e_w1, const float* __restrict__ e_b1,
                      const float* __restrict__ e_w2, const float* __restrict__ e_b2,
                      const float* __restrict__ o_w1, const float* __restrict__ o_b1,
                      const float* __restrict__ o_w2, const float* __restrict__ o_b2,
                      float* __restrict__ out)            // [B][P]
{
    __shared__ float nodes_s[PP * HID];        // 1 KB
    __shared__ float e1s[PP * PP * HID];       // 4 KB
    __shared__ float e2s[PP * PP * HID];       // 4 KB
    __shared__ float msgs[PP * HID];           // 1 KB
    __shared__ float o1s[PP * HID];            // 1 KB

    const int b   = blockIdx.x;
    const int tid = threadIdx.x;

    nodes_s[tid] = nodes[(size_t)b * PP * HID + tid];
    __syncthreads();

    // e1 = relu([node_i, node_j] @ e_w1 + e_b1)   edge e = i*4 + j
#pragma unroll
    for (int r = 0; r < 4; ++r) {
        const int idx = tid + r * 256;           // 1024 outputs
        const int e = idx >> 6, h = idx & 63;
        const int i = e >> 2, j = e & 3;
        float acc = e_b1[h];
        for (int k = 0; k < HID; ++k) {
            acc += nodes_s[i * HID + k] * e_w1[k * HID + h];
            acc += nodes_s[j * HID + k] * e_w1[(HID + k) * HID + h];
        }
        e1s[idx] = fmaxf(acc, 0.0f);
    }
    __syncthreads();

    // e2 = relu(e1 @ e_w2 + e_b2)
#pragma unroll
    for (int r = 0; r < 4; ++r) {
        const int idx = tid + r * 256;
        const int e = idx >> 6, h = idx & 63;
        float acc = e_b2[h];
        for (int k = 0; k < HID; ++k)
            acc += e1s[e * HID + k] * e_w2[k * HID + h];
        e2s[idx] = fmaxf(acc, 0.0f);
    }
    __syncthreads();

    // msg[j] = mean over send axis i
    {
        const int j = tid >> 6, h = tid & 63;
        float mres = 0.0f;
#pragma unroll
        for (int i = 0; i < PP; ++i)
            mres += e2s[(i * PP + j) * HID + h];
        msgs[j * HID + h] = mres * (1.0f / PP);
    }
    __syncthreads();

    // o1 = relu(msg @ o_w1 + o_b1)
    {
        const int j = tid >> 6, h = tid & 63;
        float acc = o_b1[h];
        for (int k = 0; k < HID; ++k)
            acc += msgs[j * HID + k] * o_w1[k * HID + h];
        o1s[j * HID + h] = fmaxf(acc, 0.0f);
    }
    __syncthreads();

    // out[b][p] = o1[p] @ o_w2 + o_b2
    if (tid < PP) {
        float acc = o_b2[0];
        for (int h = 0; h < HID; ++h)
            acc += o1s[tid * HID + h] * o_w2[h];
        out[(size_t)b * PP + tid] = acc;
    }
}

// ---------------------------------------------------------------------------
extern "C" void kernel_launch(void* const* d_in, const int* in_sizes, int n_in,
                              void* d_out, int out_size, void* d_ws, size_t ws_size,
                              hipStream_t stream) {
    const float* x      = (const float*)d_in[0];
    const float* conv_w = (const float*)d_in[1];
    const float* conv_b = (const float*)d_in[2];
    const float* e_w1   = (const float*)d_in[3];
    const float* e_b1   = (const float*)d_in[4];
    const float* e_w2   = (const float*)d_in[5];
    const float* e_b2   = (const float*)d_in[6];
    const float* o_w1   = (const float*)d_in[7];
    const float* o_b1   = (const float*)d_in[8];
    const float* o_w2   = (const float*)d_in[9];
    const float* o_b2   = (const float*)d_in[10];

    float* nodes = (float*)d_ws;               // [B][P][HID] = 512*64 floats
    float* out   = (float*)d_out;              // [B][P] = 512 floats

    conv_pool_kernel<<<NIMG, 256, 0, stream>>>(x, conv_w, conv_b, nodes);
    graph_mlp_kernel<<<BB, 256, 0, stream>>>(nodes, e_w1, e_b1, e_w2, e_b2,
                                             o_w1, o_b1, o_w2, o_b2, out);
}
